// FavorPlusAttention_42262478193312
// MI455X (gfx1250) — matrix operand order repaired and checked by reading, not compiled, not run on hardware
//
#include <hip/hip_runtime.h>

// Problem dimensions (fixed by the reference).
#define B_   2
#define T_   4096
#define D_   1024
#define H_   16
#define M_   256
#define DK_  64
#define BT_  (B_ * T_)   // 8192
#define BH_  (B_ * H_)   // 32

// LDS B-panel row pitch (bf16 elems) for the 64-wide K stage.
// 72 elems = 144 B = 36 banks; bank-quad stride (9) is coprime with 16
// -> the 16 lanes' ds_load_b128 hit 16 distinct bank quads (conflict-free),
// and 144 B keeps 16 B alignment for b128 LDS ops.
#define BP_  72

typedef __bf16 bf16;
typedef __attribute__((ext_vector_type(16))) __bf16 v16bf;
typedef __attribute__((ext_vector_type(8)))  __bf16 v8bf;
typedef __attribute__((ext_vector_type(8)))  float  v8f;

// ---------------------------------------------------------------------------
// WMMA tile loaders (wave32). Layouts used here (contiguous k, not the ISA table order; A and B match):
//  A 16x32 bf16: lanes 0-15 -> rows 0-15, K=0..15 contiguous;
//                lanes 16-31 -> same rows, K=16..31 contiguous.
//  B 32x16 bf16: lanes 0-15 -> cols, K=0..15 contiguous;
//                lanes 16-31 -> cols, K=16..31 contiguous.
//  C/D f32 16x16: lane -> col = lane&15, rows (lane>>4)*8 + i.
// ---------------------------------------------------------------------------
static __device__ __forceinline__ v16bf load_a16x32(const bf16* A, int lda) {
  const int lane = threadIdx.x & 31;
  const int row  = lane & 15;
  const int half = lane >> 4;
  const bf16* p = A + (size_t)row * lda + half * 16;
  v8bf lo = *(const v8bf*)(p);
  v8bf hi = *(const v8bf*)(p + 8);
  return __builtin_shufflevector(lo, hi, 0, 1, 2, 3, 4, 5, 6, 7,
                                 8, 9, 10, 11, 12, 13, 14, 15);
}

static __device__ __forceinline__ v16bf load_b32x16(const bf16* Bm, int ldb) {
  const int lane = threadIdx.x & 31;
  const int col  = lane & 15;
  const int half = lane >> 4;
  return *(const v16bf*)(Bm + (size_t)col * ldb + half * 16);
}

static __device__ __forceinline__ v8f wmma_bf16(v16bf a, v16bf b, v8f c) {
  return __builtin_amdgcn_wmma_f32_16x16x32_bf16(
      /*neg_a=*/false, a, /*neg_b=*/false, b,
      /*c_mod=*/(short)0, c, /*reuse_a=*/false, /*reuse_b=*/false);
}

// gfx1250 async copy: LDS[VGPR dst] <= MEM[VGPR addr64], 16 bytes/lane.
// Tracked by ASYNCcnt (CDNA5 ISA ch.10 / 15.18.3).
static __device__ __forceinline__ void async_b128_to_lds(unsigned lds_byte_off,
                                                         const void* gaddr) {
  asm volatile("global_load_async_to_lds_b128 %0, %1, off"
               :: "v"(lds_byte_off), "v"(gaddr)
               : "memory");
}

static __device__ __forceinline__ void wait_async0() {
  asm volatile("s_wait_asynccnt 0x0" ::: "memory");
}

// ---------------------------------------------------------------------------
// K0: fp32 -> bf16 conversion
// ---------------------------------------------------------------------------
__global__ void k_cvt_bf16(const float* __restrict__ in, bf16* __restrict__ out, int n) {
  int i = blockIdx.x * blockDim.x + threadIdx.x;
  if (i < n) out[i] = (bf16)in[i];
}

// ---------------------------------------------------------------------------
// K1: Y = scale * (X @ W^T), X:[BT,D] bf16, W:[D,D] bf16 (row e contiguous in k)
// B-panel (64 n-rows x 64 k per stage, 8 KB) staged in LDS via double-buffered
// async copies shared by all 8 waves; 8 WMMAs per wave per barrier.
// Store per-head layout [bh][T][dk] (transposeV=0) or [bh][dk][T] (=1, for V).
// Block 256 (8 waves); grid (BT/128, D/64).
// ---------------------------------------------------------------------------
__global__ void k_proj_gemm(const bf16* __restrict__ X, const bf16* __restrict__ W,
                            bf16* __restrict__ Y, float scale, int transposeV) {
  __shared__ bf16 sB[2][64][BP_];

  const int wave = threadIdx.x >> 5;
  const int m0 = blockIdx.x * 128 + wave * 16;
  const int n0 = blockIdx.y * 64;
  const bf16* Arow = X + (size_t)m0 * D_;

  const int tid = threadIdx.x;
  const unsigned sbase = (unsigned)(size_t)(&sB[0][0][0]);

  // Fill one 64x64 bf16 panel: 512 x 16B transfers = 256 threads x 2.
  auto fill_async = [&](int buf, int kb) {
#pragma unroll
    for (int i = 0; i < 2; ++i) {
      const int linear = tid + 256 * i;
      const int row = linear >> 3;   // 0..63 : n-local row
      const int seg = linear & 7;    // 16B segment within the 128B row
      const bf16* src = W + (size_t)(n0 + row) * D_ + kb + seg * 8;
      unsigned dst = sbase + (unsigned)(((buf * 64 + row) * BP_ + seg * 8) * 2);
      async_b128_to_lds(dst, src);
    }
  };

  v8f acc0 = {}, acc1 = {}, acc2 = {}, acc3 = {};

  fill_async(0, 0);
  wait_async0();
  __syncthreads();

  int cur = 0;
  for (int kb = 0; kb < D_; kb += 64) {
    if (kb + 64 < D_) {
      fill_async(cur ^ 1, kb + 64);
      __builtin_prefetch(Arow + kb + 64, 0, 1);
    }
    const bf16* bbase = &sB[cur][0][0];
#pragma unroll
    for (int kc = 0; kc < 64; kc += 32) {
      v16bf a  = load_a16x32(Arow + kb + kc, D_);
      v16bf b0 = load_b32x16(bbase +  0 * BP_ + kc, BP_);
      v16bf b1 = load_b32x16(bbase + 16 * BP_ + kc, BP_);
      v16bf b2 = load_b32x16(bbase + 32 * BP_ + kc, BP_);
      v16bf b3 = load_b32x16(bbase + 48 * BP_ + kc, BP_);
      acc0 = wmma_bf16(a, b0, acc0);
      acc1 = wmma_bf16(a, b1, acc1);
      acc2 = wmma_bf16(a, b2, acc2);
      acc3 = wmma_bf16(a, b3, acc3);
    }
    wait_async0();
    __syncthreads();
    cur ^= 1;
  }

  const int lane = threadIdx.x & 31;
  const int col = lane & 15;
  const int rowbase = (lane >> 4) * 8;
  v8f accs[4] = {acc0, acc1, acc2, acc3};
  for (int j = 0; j < 4; ++j) {
    for (int i = 0; i < 8; ++i) {
      int r = m0 + rowbase + i;
      int e = n0 + j * 16 + col;
      float v = accs[j][i] * scale;
      int b = r >> 12;          // r / T_
      int t = r & (T_ - 1);
      int h = e >> 6;           // e / DK_
      int dd = e & (DK_ - 1);
      size_t idx = transposeV
          ? (((size_t)(b * H_ + h) * DK_ + dd) * T_ + t)
          : (((size_t)(b * H_ + h) * T_ + t) * DK_ + dd);
      Y[idx] = (bf16)v;
    }
  }
}

// ---------------------------------------------------------------------------
// K2: FAVOR+ features. Per workgroup: 16 t-rows x full M=256 features
// (so per-row max over all M is available for the exp epilogue).
// phi = (exp(xw - rowmax) * exp(-0.5*|q|^2) + EPS) / sqrt(M)
// Block 256 (8 waves, each 2 N-tiles); grid (T/16, BH).
// transposed=0 -> out[bh][t][m] (for phiQ); =1 -> out[bh][m][t] (for phiK^T).
// ---------------------------------------------------------------------------
__global__ void k_phi(const bf16* __restrict__ Qall, const bf16* __restrict__ proj,
                      bf16* __restrict__ out, int transposed) {
  const int bh = blockIdx.y;
  const int h  = bh & (H_ - 1);
  const int t0 = blockIdx.x * 16;
  const bf16* Qrow = Qall + ((size_t)bh * T_ + t0) * DK_;
  const bf16* P = proj + (size_t)h * M_ * DK_;
  const int wave = threadIdx.x >> 5;
  const int n0 = wave * 32;

  __shared__ float s_rmax[8][16];
  __shared__ float s_qn[16];

  if (threadIdx.x < 16) {
    float s = 0.f;
    const bf16* q = Qrow + (size_t)threadIdx.x * DK_;
    for (int k = 0; k < DK_; ++k) { float f = (float)q[k]; s += f * f; }
    s_qn[threadIdx.x] = s;
  }

  v8f acc0 = {}, acc1 = {};
#pragma unroll
  for (int kb = 0; kb < DK_; kb += 32) {
    v16bf a  = load_a16x32(Qrow + kb, DK_);
    v16bf b0 = load_b32x16(P + (size_t)(n0 +  0) * DK_ + kb, DK_);
    v16bf b1 = load_b32x16(P + (size_t)(n0 + 16) * DK_ + kb, DK_);
    acc0 = wmma_bf16(a, b0, acc0);
    acc1 = wmma_bf16(a, b1, acc1);
  }

  const int lane = threadIdx.x & 31;
  const int col = lane & 15;
  const int half = lane >> 4;

  // per-lane row maxima over this wave's 32 columns
  float pm[8];
  for (int i = 0; i < 8; ++i) pm[i] = fmaxf(acc0[i], acc1[i]);
  // reduce across the 16 lanes of each half (cols)
  for (int msk = 8; msk >= 1; msk >>= 1)
    for (int i = 0; i < 8; ++i) pm[i] = fmaxf(pm[i], __shfl_xor(pm[i], msk, 32));
  if (col == 0)
    for (int i = 0; i < 8; ++i) s_rmax[wave][half * 8 + i] = pm[i];
  __syncthreads();

  const float inv_sqrt_m = 0.0625f;  // 1/sqrt(256)
  for (int i = 0; i < 8; ++i) {
    const int rowl = half * 8 + i;
    float rm = s_rmax[0][rowl];
    for (int w = 1; w < 8; ++w) rm = fmaxf(rm, s_rmax[w][rowl]);
    const float en = __expf(-0.5f * s_qn[rowl]);
    const int t = t0 + rowl;
    const float v0 = (__expf(acc0[i] - rm) * en + 1e-6f) * inv_sqrt_m;
    const float v1 = (__expf(acc1[i] - rm) * en + 1e-6f) * inv_sqrt_m;
    if (transposed) {
      out[((size_t)bh * M_ + (n0 + col)) * T_ + t]        = (bf16)v0;
      out[((size_t)bh * M_ + (n0 + 16 + col)) * T_ + t]   = (bf16)v1;
    } else {
      out[((size_t)bh * T_ + t) * M_ + (n0 + col)]        = (bf16)v0;
      out[((size_t)bh * T_ + t) * M_ + (n0 + 16 + col)]   = (bf16)v1;
    }
  }
}

// ---------------------------------------------------------------------------
// K3: Ksum[bh][m] = sum_t phiKt[bh][m][t]. One wave per row.
// ---------------------------------------------------------------------------
__global__ void k_ksum(const bf16* __restrict__ phiKt, float* __restrict__ Ksum) {
  const int wave = threadIdx.x >> 5;
  const int lane = threadIdx.x & 31;
  const int row = blockIdx.x * 8 + wave;   // [0, BH_*M_)
  const bf16* p = phiKt + (size_t)row * T_;
  float s = 0.f;
  for (int t = lane; t < T_; t += 32) s += (float)p[t];
  for (int msk = 16; msk >= 1; msk >>= 1) s += __shfl_xor(s, msk, 32);
  if (lane == 0) Ksum[row] = s;
}

// ---------------------------------------------------------------------------
// K4: KV^T[bh][d'][m] = sum_t phiKt[bh][m][t] * Vt[bh][d'][t]
// GEMM M=256(m), N=64(d'), K=4096(t). Block 128 (4 waves, one n-tile each);
// grid (M/16, BH).
// ---------------------------------------------------------------------------
__global__ void k_kv(const bf16* __restrict__ phiKt, const bf16* __restrict__ Vt,
                     bf16* __restrict__ KVt) {
  const int bh = blockIdx.y;
  const int m0 = blockIdx.x * 16;
  const int wave = threadIdx.x >> 5;
  const int n0 = wave * 16;
  const bf16* A  = phiKt + ((size_t)bh * M_ + m0) * T_;
  const bf16* Bv = Vt + ((size_t)bh * DK_ + n0) * T_;
  v8f acc = {};
#pragma unroll 4
  for (int kb = 0; kb < T_; kb += 32) {
    if (kb + 32 < T_) __builtin_prefetch(A + kb + 32, 0, 1);
    v16bf a = load_a16x32(A + kb, T_);
    v16bf b = load_b32x16(Bv + kb, T_);
    acc = wmma_bf16(a, b, acc);
  }
  const int lane = threadIdx.x & 31;
  const int col = lane & 15;
  const int rowbase = (lane >> 4) * 8;
  for (int i = 0; i < 8; ++i) {
    int m = m0 + rowbase + i;
    int dd = n0 + col;
    KVt[((size_t)bh * DK_ + dd) * M_ + m] = (bf16)acc[i];
  }
}

// ---------------------------------------------------------------------------
// K5: num = phiQ @ KV, den = phiQ . Ksum + EPS; merged = num/den in [B,T,D].
// Block 128 (4 waves, one n-tile each); grid (T/16, BH).
// ---------------------------------------------------------------------------
__global__ void k_numden(const bf16* __restrict__ phiQ, const bf16* __restrict__ KVt,
                         const float* __restrict__ Ksum, bf16* __restrict__ merged) {
  const int bh = blockIdx.y;
  const int b = bh >> 4;
  const int h = bh & (H_ - 1);
  const int t0 = blockIdx.x * 16;
  const int wave = threadIdx.x >> 5;
  const int n0 = wave * 16;
  const bf16* A  = phiQ + ((size_t)bh * T_ + t0) * M_;
  const bf16* Bk = KVt + ((size_t)bh * DK_ + n0) * M_;

  __shared__ float s_den[16];
  if (threadIdx.x < 16) {
    const bf16* a = A + (size_t)threadIdx.x * M_;
    const float* ks = Ksum + (size_t)bh * M_;
    float s = 0.f;
    for (int m = 0; m < M_; ++m) s += (float)a[m] * ks[m];
    s_den[threadIdx.x] = s + 1e-6f;
  }

  v8f acc = {};
#pragma unroll
  for (int kb = 0; kb < M_; kb += 32) {
    v16bf a  = load_a16x32(A + kb, M_);
    v16bf bm = load_b32x16(Bk + kb, M_);
    acc = wmma_bf16(a, bm, acc);
  }
  __syncthreads();

  const int lane = threadIdx.x & 31;
  const int col = lane & 15;
  const int rowbase = (lane >> 4) * 8;
  for (int i = 0; i < 8; ++i) {
    int rowl = rowbase + i;
    int t = t0 + rowl;
    int e = h * DK_ + n0 + col;
    float v = acc[i] / s_den[rowl];
    merged[((size_t)b * T_ + t) * D_ + e] = (bf16)v;
  }
}

// ---------------------------------------------------------------------------
// K6: out = merged @ Wo^T + bo, fp32 output. Same async-LDS staging as K1.
// ---------------------------------------------------------------------------
__global__ void k_out_gemm(const bf16* __restrict__ Xm, const bf16* __restrict__ W,
                           const float* __restrict__ bo, float* __restrict__ out) {
  __shared__ bf16 sB[2][64][BP_];

  const int wave = threadIdx.x >> 5;
  const int m0 = blockIdx.x * 128 + wave * 16;
  const int n0 = blockIdx.y * 64;
  const bf16* Arow = Xm + (size_t)m0 * D_;

  const int tid = threadIdx.x;
  const unsigned sbase = (unsigned)(size_t)(&sB[0][0][0]);

  auto fill_async = [&](int buf, int kb) {
#pragma unroll
    for (int i = 0; i < 2; ++i) {
      const int linear = tid + 256 * i;
      const int row = linear >> 3;
      const int seg = linear & 7;
      const bf16* src = W + (size_t)(n0 + row) * D_ + kb + seg * 8;
      unsigned dst = sbase + (unsigned)(((buf * 64 + row) * BP_ + seg * 8) * 2);
      async_b128_to_lds(dst, src);
    }
  };

  v8f acc0 = {}, acc1 = {}, acc2 = {}, acc3 = {};

  fill_async(0, 0);
  wait_async0();
  __syncthreads();

  int cur = 0;
  for (int kb = 0; kb < D_; kb += 64) {
    if (kb + 64 < D_) {
      fill_async(cur ^ 1, kb + 64);
      __builtin_prefetch(Arow + kb + 64, 0, 1);
    }
    const bf16* bbase = &sB[cur][0][0];
#pragma unroll
    for (int kc = 0; kc < 64; kc += 32) {
      v16bf a  = load_a16x32(Arow + kb + kc, D_);
      v16bf b0 = load_b32x16(bbase +  0 * BP_ + kc, BP_);
      v16bf b1 = load_b32x16(bbase + 16 * BP_ + kc, BP_);
      v16bf b2 = load_b32x16(bbase + 32 * BP_ + kc, BP_);
      v16bf b3 = load_b32x16(bbase + 48 * BP_ + kc, BP_);
      acc0 = wmma_bf16(a, b0, acc0);
      acc1 = wmma_bf16(a, b1, acc1);
      acc2 = wmma_bf16(a, b2, acc2);
      acc3 = wmma_bf16(a, b3, acc3);
    }
    wait_async0();
    __syncthreads();
    cur ^= 1;
  }

  const int lane = threadIdx.x & 31;
  const int col = lane & 15;
  const int rowbase = (lane >> 4) * 8;
  v8f accs[4] = {acc0, acc1, acc2, acc3};
  for (int j = 0; j < 4; ++j) {
    for (int i = 0; i < 8; ++i) {
      int r = m0 + rowbase + i;
      int e = n0 + j * 16 + col;
      out[(size_t)r * D_ + e] = accs[j][i] + bo[e];
    }
  }
}

// ---------------------------------------------------------------------------
// Host-side launch
// ---------------------------------------------------------------------------
extern "C" void kernel_launch(void* const* d_in, const int* in_sizes, int n_in,
                              void* d_out, int out_size, void* d_ws, size_t ws_size,
                              hipStream_t stream) {
  (void)in_sizes; (void)n_in; (void)out_size; (void)ws_size;
  const float* x    = (const float*)d_in[0];
  const float* Wq   = (const float*)d_in[1];
  const float* Wk   = (const float*)d_in[2];
  const float* Wv   = (const float*)d_in[3];
  const float* Wo   = (const float*)d_in[4];
  const float* bo   = (const float*)d_in[5];
  const float* proj = (const float*)d_in[6];

  char* ws = (char*)d_ws;
  size_t off = 0;
  auto take = [&](size_t bytes) -> char* {
    char* p = ws + off;
    off = (off + bytes + 255) & ~(size_t)255;
    return p;
  };
  bf16* xb    = (bf16*)take((size_t)BT_ * D_ * 2);
  bf16* wqb   = (bf16*)take((size_t)D_ * D_ * 2);
  bf16* wkb   = (bf16*)take((size_t)D_ * D_ * 2);
  bf16* wvb   = (bf16*)take((size_t)D_ * D_ * 2);
  bf16* wob   = (bf16*)take((size_t)D_ * D_ * 2);
  bf16* projb = (bf16*)take((size_t)H_ * M_ * DK_ * 2);
  bf16* Qb    = (bf16*)take((size_t)BH_ * T_ * DK_ * 2);
  bf16* Kb    = (bf16*)take((size_t)BH_ * T_ * DK_ * 2);
  bf16* Vtb   = (bf16*)take((size_t)BH_ * DK_ * T_ * 2);
  bf16* phiQ  = (bf16*)take((size_t)BH_ * T_ * M_ * 2);
  bf16* phiKt = (bf16*)take((size_t)BH_ * M_ * T_ * 2);
  bf16* KVt   = (bf16*)take((size_t)BH_ * DK_ * M_ * 2);
  bf16* merged= (bf16*)take((size_t)BT_ * D_ * 2);
  float* Ksum = (float*)take((size_t)BH_ * M_ * 4);

  auto cvt = [&](const float* src, bf16* dst, int n) {
    k_cvt_bf16<<<(n + 255) / 256, 256, 0, stream>>>(src, dst, n);
  };
  cvt(x, xb, BT_ * D_);
  cvt(Wq, wqb, D_ * D_);
  cvt(Wk, wkb, D_ * D_);
  cvt(Wv, wvb, D_ * D_);
  cvt(Wo, wob, D_ * D_);
  cvt(proj, projb, H_ * M_ * DK_);

  const float inv_sqrt_dk = 0.125f;  // 1/sqrt(64)
  dim3 gp(BT_ / 128, D_ / 64);       // (64, 16)
  k_proj_gemm<<<gp, 256, 0, stream>>>(xb, wqb, Qb,  inv_sqrt_dk, 0);
  k_proj_gemm<<<gp, 256, 0, stream>>>(xb, wkb, Kb,  inv_sqrt_dk, 0);
  k_proj_gemm<<<gp, 256, 0, stream>>>(xb, wvb, Vtb, 1.0f,        1);

  dim3 gphi(T_ / 16, BH_);           // (256, 32)
  k_phi<<<gphi, 256, 0, stream>>>(Qb, projb, phiQ, 0);
  k_phi<<<gphi, 256, 0, stream>>>(Kb, projb, phiKt, 1);

  k_ksum<<<(BH_ * M_) / 8, 256, 0, stream>>>(phiKt, Ksum);

  dim3 gkv(M_ / 16, BH_);            // (16, 32)
  k_kv<<<gkv, 128, 0, stream>>>(phiKt, Vtb, KVt);

  dim3 gnum(T_ / 16, BH_);           // (256, 32)
  k_numden<<<gnum, 128, 0, stream>>>(phiQ, KVt, Ksum, merged);

  k_out_gemm<<<gp, 256, 0, stream>>>(merged, wob, bo, (float*)d_out);
}
